// LightGODE_71116068487678
// MI455X (gfx1250) — matrix operand order repaired and checked by reading, not compiled, not run on hardware
//
#include <hip/hip_runtime.h>
#include <math.h>

#define NUM_USERS 100000
#define NUM_ITEMS 50000
#define NTOT (NUM_USERS + NUM_ITEMS)
#define EMB 64
#define NNZ_C 4000000
#define BSZ 4096
#define REG_LAMBDA_C 1.0e-4
#define EPS_C 1e-12f

typedef _Float16 half_t;
typedef half_t v16h __attribute__((ext_vector_type(16)));
typedef float  v8f  __attribute__((ext_vector_type(8)));

// ---- workspace layout (bytes) ----
static constexpr size_t OFF_MAP  = 0;                                   // NTOT int32 = 600000
static constexpr size_t OFF_AXC  = 600064;                              // 2B*64 f32 = 2097152
static constexpr size_t OFF_UHI  = OFF_AXC + (size_t)2*BSZ*EMB*4;       // B*64 half each
static constexpr size_t OFF_ULO  = OFF_UHI + (size_t)BSZ*EMB*2;
static constexpr size_t OFF_IHI  = OFF_ULO + (size_t)BSZ*EMB*2;
static constexpr size_t OFF_ILO  = OFF_IHI + (size_t)BSZ*EMB*2;
static constexpr size_t OFF_PU   = OFF_ILO + (size_t)BSZ*EMB*2;         // 4096 doubles each
static constexpr size_t OFF_PI   = OFF_PU + (size_t)4096*8;
static constexpr size_t OFF_PA   = OFF_PI + (size_t)4096*8;
static constexpr size_t OFF_PR   = OFF_PA + (size_t)4096*8;             // end ~4.93 MB

__device__ __forceinline__ float wave_sum(float v) {
  #pragma unroll
  for (int m = 16; m > 0; m >>= 1) v += __shfl_xor(v, m, 32);
  return v;
}

// ---- 1) init map + compact-ax (partials need no init: fully overwritten) ----
__global__ void k_init(int* __restrict__ map, float* __restrict__ axc) {
  int t = blockIdx.x * blockDim.x + threadIdx.x;
  if (t < NTOT) map[t] = -1;
  if (t < 2 * BSZ * EMB) axc[t] = 0.0f;
}

// ---- 2) row -> compact slot map (first writer wins; duplicates share slot) ----
__global__ void k_build_map(const int* __restrict__ user, const int* __restrict__ positive,
                            int* __restrict__ map) {
  int t = blockIdx.x * blockDim.x + threadIdx.x;
  if (t >= 2 * BSZ) return;
  int row = (t < BSZ) ? user[t] : (NUM_USERS + positive[t - BSZ]);
  atomicCAS(&map[row], -1, t);
}

// ---- 3) sparse scatter, restricted to needed rows (~5% hit rate; map is L2-resident) ----
__global__ void k_scatter(const int* __restrict__ rows, const int* __restrict__ cols,
                          const float* __restrict__ vals, const int* __restrict__ map,
                          const float* __restrict__ uemb, const float* __restrict__ iemb,
                          float* __restrict__ axc) {
  int e = blockIdx.x * blockDim.x + threadIdx.x;
  if (e >= NNZ_C) return;
  int s = map[rows[e]];
  if (s < 0) return;
  int c = cols[e];
  float v = vals[e];
  const float* xc = (c < NUM_USERS) ? (uemb + (size_t)c * EMB)
                                    : (iemb + (size_t)(c - NUM_USERS) * EMB);
  float* dst = axc + (size_t)s * EMB;
  #pragma unroll
  for (int d = 0; d < EMB; ++d) atomicAdd(dst + d, v * xc[d]);
}

// ---- 4) gather z = 2x + ax, normalize, emit f16 hi/lo, per-wave align/reg partials ----
__global__ void __launch_bounds__(256)
k_gather(const int* __restrict__ user, const int* __restrict__ positive,
         const int* __restrict__ map, const float* __restrict__ axc,
         const float* __restrict__ uemb, const float* __restrict__ iemb,
         half_t* __restrict__ uhi, half_t* __restrict__ ulo,
         half_t* __restrict__ ihi, half_t* __restrict__ ilo,
         double* __restrict__ partA, double* __restrict__ partR) {
  int wid  = (blockIdx.x * blockDim.x + threadIdx.x) >> 5;
  int lane = threadIdx.x & 31;
  if (wid >= BSZ) return;
  int d0 = lane, d1 = lane + 32;

  int ru = user[wid];
  int su = map[ru];
  float e0 = uemb[(size_t)ru * EMB + d0], e1 = uemb[(size_t)ru * EMB + d1];
  float z0 = 2.0f * e0 + axc[(size_t)su * EMB + d0];
  float z1 = 2.0f * e1 + axc[(size_t)su * EMB + d1];
  float inv = rsqrtf(wave_sum(z0 * z0 + z1 * z1) + EPS_C);
  float un0 = z0 * inv, un1 = z1 * inv;

  int ri = positive[wid];
  int si = map[NUM_USERS + ri];
  float f0 = iemb[(size_t)ri * EMB + d0], f1 = iemb[(size_t)ri * EMB + d1];
  float w0 = 2.0f * f0 + axc[(size_t)si * EMB + d0];
  float w1 = 2.0f * f1 + axc[(size_t)si * EMB + d1];
  float invi = rsqrtf(wave_sum(w0 * w0 + w1 * w1) + EPS_C);
  float in0 = w0 * invi, in1 = w1 * invi;

  size_t o0 = (size_t)wid * EMB + d0, o1 = (size_t)wid * EMB + d1;
  half_t h;
  h = (half_t)un0; uhi[o0] = h; ulo[o0] = (half_t)(un0 - (float)h);
  h = (half_t)un1; uhi[o1] = h; ulo[o1] = (half_t)(un1 - (float)h);
  h = (half_t)in0; ihi[o0] = h; ilo[o0] = (half_t)(in0 - (float)h);
  h = (half_t)in1; ihi[o1] = h; ilo[o1] = (half_t)(in1 - (float)h);

  float du0 = un0 - in0, du1 = un1 - in1;
  float dsq  = wave_sum(du0 * du0 + du1 * du1);
  float regp = wave_sum(e0 * e0 + e1 * e1 + f0 * f0 + f1 * f1);
  if (lane == 0) {          // one slot per wave: no atomics, deterministic
    partA[wid] = (double)dsq;
    partR[wid] = (double)regp;
  }
}

// ---- 5) upper-triangular Gram tiles via v_wmma_f32_16x16x32_f16 (hi/lo split).
// Wave handles 4 tile-columns of one tile-row, reusing the A fragments.
// Off-diag tile: full 16x16 sum counts each pair once. Diag tile: (sum-16)/2.
__global__ void __launch_bounds__(128)
k_gram(const half_t* __restrict__ hi, const half_t* __restrict__ lo,
       double* __restrict__ partG) {
  int wave = threadIdx.x >> 5;
  int lane = threadIdx.x & 31;
  int tr = blockIdx.y;                          // tile row 0..255
  int tcBase = (blockIdx.x * 4 + wave) * 4;     // 4 tile cols per wave
  int mlane = lane & 15, halfid = lane >> 4;

  double wsum = 0.0;
  if (tcBase + 3 >= tr) {                       // wave has at least one live tile
    v16h ah[2], al[2];                          // A fragments, both K-steps
    int arow = tr * 16 + mlane;
    #pragma unroll
    for (int ks = 0; ks < 2; ++ks) {
      #pragma unroll
      for (int vg = 0; vg < 8; ++vg) {          // A 16x32 f16 layout (ISA 7.12.2)
        int k = ks * 32 + ((vg >> 2) << 4) + halfid * 8 + ((vg & 3) << 1);
        size_t ia = (size_t)arow * EMB + k;
        ah[ks][2 * vg] = hi[ia]; ah[ks][2 * vg + 1] = hi[ia + 1];
        al[ks][2 * vg] = lo[ia]; al[ks][2 * vg + 1] = lo[ia + 1];
      }
    }
    for (int t = 0; t < 4; ++t) {
      int tc = tcBase + t;
      if (tc < tr) continue;
      v8f acc = {};
      int brow = tc * 16 + mlane;               // B[k][n] = X[tc*16+n][k]
      #pragma unroll
      for (int ks = 0; ks < 2; ++ks) {
        v16h bh, bl;
        #pragma unroll
        for (int vg = 0; vg < 8; ++vg) {
          int k = ks * 32 + ((vg >> 2) << 4) + halfid * 8 + ((vg & 3) << 1);
          size_t ib = (size_t)brow * EMB + k;
          bh[2 * vg] = hi[ib]; bh[2 * vg + 1] = hi[ib + 1];
          bl[2 * vg] = lo[ib]; bl[2 * vg + 1] = lo[ib + 1];
        }
        // G = hi*hi + hi*lo + lo*hi (drop lo*lo): ~f32-accurate unit-vector dots
        acc = __builtin_amdgcn_wmma_f32_16x16x32_f16(false, ah[ks], false, bh, (short)0, acc, false, false);
        acc = __builtin_amdgcn_wmma_f32_16x16x32_f16(false, ah[ks], false, bl, (short)0, acc, false, false);
        acc = __builtin_amdgcn_wmma_f32_16x16x32_f16(false, al[ks], false, bh, (short)0, acc, false, false);
      }
      float part = 0.0f;
      #pragma unroll
      for (int r = 0; r < 8; ++r) {
        float g  = acc[r];
        float sq = fmaxf(2.0f - 2.0f * g, 0.0f);
        part += __expf(-2.0f * sq);
      }
      part = wave_sum(part);
      wsum += (tc == tr) ? ((double)part - 16.0) * 0.5 : (double)part;
    }
  }
  __shared__ double sred[4];
  if (lane == 0) sred[wave] = wsum;
  __syncthreads();
  if (threadIdx.x == 0)                         // one slot per block: no atomics
    partG[blockIdx.y * 16 + blockIdx.x] = sred[0] + sred[1] + sred[2] + sred[3];
}

// ---- 6) parallel fold of all partials + finalize three scalars ----
__global__ void __launch_bounds__(256)
k_final(const double* __restrict__ partU, const double* __restrict__ partI,
        const double* __restrict__ partA, const double* __restrict__ partR,
        float* __restrict__ out) {
  int t = threadIdx.x;
  double su = 0, si = 0, sa = 0, sr = 0;
  for (int j = t; j < 4096; j += 256) {
    su += partU[j]; si += partI[j]; sa += partA[j]; sr += partR[j];
  }
  #pragma unroll
  for (int m = 16; m > 0; m >>= 1) {
    su += __shfl_xor(su, m, 32); si += __shfl_xor(si, m, 32);
    sa += __shfl_xor(sa, m, 32); sr += __shfl_xor(sr, m, 32);
  }
  __shared__ double red[4][8];
  int w = t >> 5;
  if ((t & 31) == 0) { red[0][w] = su; red[1][w] = si; red[2][w] = sa; red[3][w] = sr; }
  __syncthreads();
  if (t == 0) {
    double SU = 0, SI = 0, SA = 0, SR = 0;
    for (int k2 = 0; k2 < 8; ++k2) { SU += red[0][k2]; SI += red[1][k2]; SA += red[2][k2]; SR += red[3][k2]; }
    double npairs = (double)BSZ * (double)(BSZ - 1) * 0.5;
    out[0] = (float)(SA / (double)BSZ);
    out[1] = (float)(0.5 * (log(SU / npairs) + log(SI / npairs)));   // GAMMA = 1
    out[2] = (float)(REG_LAMBDA_C * 0.5 * SR / (double)BSZ);
  }
}

extern "C" void kernel_launch(void* const* d_in, const int* in_sizes, int n_in,
                              void* d_out, int out_size, void* d_ws, size_t ws_size,
                              hipStream_t stream) {
  const int*   user     = (const int*)d_in[0];
  const int*   positive = (const int*)d_in[1];
  // d_in[2] = negative (unused by reference)
  const int*   adj_rows = (const int*)d_in[3];
  const int*   adj_cols = (const int*)d_in[4];
  const float* adj_vals = (const float*)d_in[5];
  const float* user_emb = (const float*)d_in[6];
  const float* item_emb = (const float*)d_in[7];
  float* out = (float*)d_out;

  char* ws = (char*)d_ws;
  int*    map   = (int*)(ws + OFF_MAP);
  float*  axc   = (float*)(ws + OFF_AXC);
  half_t* uhi   = (half_t*)(ws + OFF_UHI);
  half_t* ulo   = (half_t*)(ws + OFF_ULO);
  half_t* ihi   = (half_t*)(ws + OFF_IHI);
  half_t* ilo   = (half_t*)(ws + OFF_ILO);
  double* partU = (double*)(ws + OFF_PU);
  double* partI = (double*)(ws + OFF_PI);
  double* partA = (double*)(ws + OFF_PA);
  double* partR = (double*)(ws + OFF_PR);

  {
    int n = 2 * BSZ * EMB;               // covers NTOT too
    k_init<<<(n + 255) / 256, 256, 0, stream>>>(map, axc);
  }
  k_build_map<<<(2 * BSZ + 255) / 256, 256, 0, stream>>>(user, positive, map);
  k_scatter<<<(NNZ_C + 255) / 256, 256, 0, stream>>>(adj_rows, adj_cols, adj_vals, map,
                                                     user_emb, item_emb, axc);
  k_gather<<<(BSZ * 32 + 255) / 256, 256, 0, stream>>>(user, positive, map, axc,
                                                       user_emb, item_emb,
                                                       uhi, ulo, ihi, ilo, partA, partR);
  dim3 gg(16, BSZ / 16, 1);              // 16 x 256 blocks, 4 waves x 4 tiles each
  k_gram<<<gg, 128, 0, stream>>>(uhi, ulo, partU);
  k_gram<<<gg, 128, 0, stream>>>(ihi, ilo, partI);
  k_final<<<1, 256, 0, stream>>>(partU, partI, partA, partR, out);
}
